// TransformerBlock_63788854280324
// MI455X (gfx1250) — compile-verified, matrix operand order checked by reading, not run on hardware
//
#include <hip/hip_runtime.h>
#include <hip/hip_bf16.h>
#include <math.h>

// ---------------------------------------------------------------------------
// MI455X (gfx1250) transformer block.
// Compute-bound (~137 GFLOP vs ~85MB unique traffic) -> all GEMMs on
// v_wmma_f32_16x16x32_bf16 (wave32), GEMM tile staging via the Tensor Data
// Mover (tensor_load_to_lds + s_wait_tensorcnt), weights pre-transposed to
// [N][K] bf16 so both operand tiles are rectangular row-major copies.
// TDM pad feature (16dw interval / 4dw pad) reproduces the 40-element padded
// LDS rows the WMMA fragment loads expect.
// ---------------------------------------------------------------------------

typedef __bf16 bf16;
typedef __attribute__((ext_vector_type(16))) __bf16 v16bf;
typedef __attribute__((ext_vector_type(8)))  __bf16 v8bf;
typedef __attribute__((ext_vector_type(4)))  __bf16 v4bf;
typedef __attribute__((ext_vector_type(8)))  float  v8f;
typedef __attribute__((ext_vector_type(4)))  unsigned u32x4;
typedef __attribute__((ext_vector_type(8)))  int      i32x8;
typedef __attribute__((ext_vector_type(4)))  int      i32x4;

static __device__ __forceinline__ bf16 f2bf(float f) { return (bf16)f; }

static __device__ __forceinline__ v8f wmma_bf16(v8bf alo, v8bf ahi,
                                                v8bf blo, v8bf bhi, v8f c) {
    v16bf a = __builtin_shufflevector(alo, ahi, 0,1,2,3,4,5,6,7,8,9,10,11,12,13,14,15);
    v16bf b = __builtin_shufflevector(blo, bhi, 0,1,2,3,4,5,6,7,8,9,10,11,12,13,14,15);
    return __builtin_amdgcn_wmma_f32_16x16x32_bf16(false, a, false, b,
                                                   (short)0, c, false, false);
}

// TDM: load a [128 rows x 32 cols] bf16 tile from row-major global (row
// stride K elements) into LDS with 4-DWORD padding per 16-DWORD row
// (-> LDS row stride 40 bf16). D# per CDNA5 ISA ch.8.
static __device__ __forceinline__ void tdm_load_tile(unsigned lds_off,
                                                     const bf16* gsrc,
                                                     int K, int rows_total) {
    unsigned long long ga = (unsigned long long)(uintptr_t)gsrc;
    u32x4 g0;
    g0[0] = 1u;                                        // count=1, user mode
    g0[1] = lds_off;                                   // lds_addr (bytes)
    g0[2] = (unsigned)ga;                              // global_addr lo
    g0[3] = (unsigned)((ga >> 32) & 0x1FFFFFFull) | (2u << 30);  // hi | type=2
    i32x8 g1;
    g1[0] = (int)((1u << 16)      // data_size = 1 -> 2 bytes
                | (1u << 20)      // pad_enable
                | (3u << 22)      // pad_interval: 16 DWORDs
                | (3u << 25));    // pad_amount: 4 DWORDs
    g1[1] = (int)(((unsigned)K & 0xFFFFu) << 16);                  // tensor_dim0 lo16
    g1[2] = (int)(((unsigned)K >> 16) | (((unsigned)rows_total & 0xFFFFu) << 16));
    g1[3] = (int)(((unsigned)rows_total >> 16) | (32u << 16));     // tile_dim0 = 32
    g1[4] = 128;                                                   // tile_dim1 = 128
    g1[5] = K;                                                     // dim0 stride lo
    g1[6] = 0;
    g1[7] = 0;
    i32x4 z4 = {0, 0, 0, 0};
#if __clang_major__ >= 23
    i32x8 z8 = {0, 0, 0, 0, 0, 0, 0, 0};
    __builtin_amdgcn_tensor_load_to_lds(g0, g1, z4, z4, z8, 0);
#else
    __builtin_amdgcn_tensor_load_to_lds(g0, g1, z4, z4, 0);
#endif
}

// ------------------------------ fp32 -> bf16 --------------------------------
__global__ __launch_bounds__(256) void cvt_f32_bf16(const float* __restrict__ in,
                                                    bf16* __restrict__ out, int n4) {
    for (int i = blockIdx.x * blockDim.x + threadIdx.x; i < n4;
         i += gridDim.x * blockDim.x) {
        float4 v = ((const float4*)in)[i];
        v4bf o; o[0]=f2bf(v.x); o[1]=f2bf(v.y); o[2]=f2bf(v.z); o[3]=f2bf(v.w);
        *(v4bf*)(out + (size_t)i * 4) = o;
    }
}

// -------------------- fp32 [K][N] -> bf16 transposed [N][K] -----------------
__global__ __launch_bounds__(256) void cvt_transpose_bf16(const float* __restrict__ W,
                                                          bf16* __restrict__ WT,
                                                          int K, int N) {
    __shared__ float t[32][33];
    const int n0 = blockIdx.x * 32, k0 = blockIdx.y * 32;
    const int tx = threadIdx.x & 31, tg = threadIdx.x >> 5;   // 32 x 8
    #pragma unroll
    for (int j = 0; j < 4; ++j) {
        int r = tg * 4 + j;
        t[r][tx] = W[(size_t)(k0 + r) * N + n0 + tx];
    }
    __syncthreads();
    #pragma unroll
    for (int j = 0; j < 4; ++j) {
        int r = tg * 4 + j;
        WT[(size_t)(n0 + r) * K + k0 + tx] = f2bf(t[tx][r]);
    }
}

// ------------------------------ LayerNorm -----------------------------------
__global__ __launch_bounds__(256) void layernorm_bf16(const float* __restrict__ x,
                                                      const float* __restrict__ scale,
                                                      const float* __restrict__ bias,
                                                      bf16* __restrict__ y) {
    const int row = blockIdx.x, t = threadIdx.x;
    __shared__ float r1[256], r2[256];
    float4 v = *(const float4*)(x + (size_t)row * 1024 + t * 4);
    float s  = v.x + v.y + v.z + v.w;
    float ss = v.x*v.x + v.y*v.y + v.z*v.z + v.w*v.w;
    r1[t] = s; r2[t] = ss; __syncthreads();
    for (int off = 128; off > 0; off >>= 1) {
        if (t < off) { r1[t] += r1[t + off]; r2[t] += r2[t + off]; }
        __syncthreads();
    }
    float mu  = r1[0] * (1.0f / 1024.0f);
    float var = r2[0] * (1.0f / 1024.0f) - mu * mu;
    float rs  = rsqrtf(var + 1e-5f);
    float4 sc = *(const float4*)(scale + t * 4);
    float4 bi = *(const float4*)(bias  + t * 4);
    v4bf o;
    o[0] = f2bf((v.x - mu) * rs * sc.x + bi.x);
    o[1] = f2bf((v.y - mu) * rs * sc.y + bi.y);
    o[2] = f2bf((v.z - mu) * rs * sc.z + bi.z);
    o[3] = f2bf((v.w - mu) * rs * sc.w + bi.w);
    *(v4bf*)(y + (size_t)row * 1024 + t * 4) = o;
}

// ------------------------------ WMMA GEMM -----------------------------------
// C[M,N] = A[M,K](bf16) x BT[N,K](bf16)^T + bias (+res) (gelu) -> f32|bf16
// 256 thr = 8 waves (2x4); wave tile 64x32 = 4x2 wmma accums; k-step 32.
// Tiles staged by the Tensor Data Mover (wave 0 issues, TENSORcnt drains).
template <int GELU, int RES, int OUTBF16>
__global__ __launch_bounds__(256) void gemm_wmma(const bf16* __restrict__ A,
                                                 const bf16* __restrict__ BT,
                                                 const float* __restrict__ bias,
                                                 const float* __restrict__ res,
                                                 void* __restrict__ out,
                                                 int M, int N, int K) {
    __shared__ bf16 As[128][40];   // [m][k], TDM-padded rows (80B stride)
    __shared__ bf16 Bs[128][40];   // [n][k]
    const int tid  = threadIdx.x;
    const int lane = tid & 31, wave = tid >> 5;
    const int wm = wave >> 2, wn = wave & 3;           // 2 x 4 wave grid
    const int m0 = blockIdx.y * 128, n0 = blockIdx.x * 128;
    const int lrow = lane & 15, kb = (lane >> 4) * 8;  // 16-bit A/B frag layout

    const unsigned lds_as = (unsigned)(uintptr_t)&As[0][0];
    const unsigned lds_bs = (unsigned)(uintptr_t)&Bs[0][0];

    v8f acc[4][2] = {};

    for (int kk = 0; kk < K; kk += 32) {
        if (wave == 0) {
            tdm_load_tile(lds_as, A  + (size_t)m0 * K + kk, K, M);
            tdm_load_tile(lds_bs, BT + (size_t)n0 * K + kk, K, N);
            __builtin_amdgcn_s_wait_tensorcnt(0);
        }
        __syncthreads();

        v8bf alo[4], ahi[4], blo[2], bhi[2];
        #pragma unroll
        for (int tm = 0; tm < 4; ++tm) {
            int r = wm * 64 + tm * 16 + lrow;
            alo[tm] = *(const v8bf*)&As[r][kb];
            ahi[tm] = *(const v8bf*)&As[r][kb + 16];
        }
        #pragma unroll
        for (int tn = 0; tn < 2; ++tn) {
            int c = wn * 32 + tn * 16 + lrow;
            blo[tn] = *(const v8bf*)&Bs[c][kb];
            bhi[tn] = *(const v8bf*)&Bs[c][kb + 16];
        }
        #pragma unroll
        for (int tm = 0; tm < 4; ++tm)
            #pragma unroll
            for (int tn = 0; tn < 2; ++tn)
                acc[tm][tn] = wmma_bf16(alo[tm], ahi[tm], blo[tn], bhi[tn], acc[tm][tn]);
        __syncthreads();
    }

    // epilogue: D-layout — VGPR i: row = i + 8*(lane>=16), col = lane%16
    #pragma unroll
    for (int tm = 0; tm < 4; ++tm)
        #pragma unroll
        for (int tn = 0; tn < 2; ++tn)
            #pragma unroll
            for (int i = 0; i < 8; ++i) {
                int row = m0 + wm * 64 + tm * 16 + i + 8 * (lane >> 4);
                int col = n0 + wn * 32 + tn * 16 + lrow;
                float v = acc[tm][tn][i] + bias[col];
                if (RES)  v += res[(size_t)row * N + col];
                if (GELU) v = 0.5f * v * (1.0f + erff(v * 0.70710678118654752f));
                if (OUTBF16) ((bf16*)out)[(size_t)row * N + col] = f2bf(v);
                else         ((float*)out)[(size_t)row * N + col] = v;
            }
}

// ------------------------------ Attention -----------------------------------
// One block per (16 query rows, head, batch). Flash-style over 128-key blocks:
// S = (Q/8)·K^T via WMMA, online softmax, O += P·V via WMMA (acc rescaled).
__global__ __launch_bounds__(256) void attention_wmma(const float* __restrict__ qkv,
                                                      bf16* __restrict__ o) {
    const int tid  = threadIdx.x;
    const int lane = tid & 31, wave = tid >> 5;
    const int lrow = lane & 15, kb = (lane >> 4) * 8;
    const int s0 = blockIdx.x * 16;
    const int bi = blockIdx.y >> 4, hi = blockIdx.y & 15;
    const size_t rowstride = 3072;                 // qkv inner dim
    const size_t base = (size_t)bi * 2048 * rowstride;
    const int qoff = hi * 64, koff = 1024 + hi * 64, voff = 2048 + hi * 64;

    __shared__ float Ssh[16][132];
    __shared__ bf16  Qs[16][64];
    __shared__ bf16  Ks[128][64];    // [key][hd]
    __shared__ bf16  Vt[64][136];    // [hd][key]
    __shared__ bf16  Ps[16][136];
    __shared__ float mrow[16], lrow_s[16], alpha_s[16];

    if (tid < 16) { mrow[tid] = -INFINITY; lrow_s[tid] = 0.0f; }
    { // stage Q, pre-scaled by 1/sqrt(64)=0.125 (exact in bf16)
        int idx = tid * 4, r = idx >> 6, c = idx & 63;
        float4 v = *(const float4*)(qkv + base + (size_t)(s0 + r) * rowstride + qoff + c);
        v4bf q; q[0]=f2bf(v.x*0.125f); q[1]=f2bf(v.y*0.125f);
        q[2]=f2bf(v.z*0.125f); q[3]=f2bf(v.w*0.125f);
        *(v4bf*)&Qs[r][c] = q;
    }
    __syncthreads();

    v8f acc = {};   // waves 0..3: 16x16 O tile at cols wave*16

    for (int kb0 = 0; kb0 < 2048; kb0 += 128) {
        { // stage K rows and V (transposed)
            int key = tid >> 1, c0 = (tid & 1) * 32;
            const float* kp = qkv + base + (size_t)(kb0 + key) * rowstride + koff + c0;
            const float* vp = qkv + base + (size_t)(kb0 + key) * rowstride + voff + c0;
            #pragma unroll
            for (int j = 0; j < 8; ++j) {
                float4 kv = *(const float4*)(kp + j * 4);
                v4bf kq; kq[0]=f2bf(kv.x); kq[1]=f2bf(kv.y); kq[2]=f2bf(kv.z); kq[3]=f2bf(kv.w);
                *(v4bf*)&Ks[key][c0 + j * 4] = kq;
                float4 vv = *(const float4*)(vp + j * 4);
                Vt[c0 + j*4 + 0][key] = f2bf(vv.x);
                Vt[c0 + j*4 + 1][key] = f2bf(vv.y);
                Vt[c0 + j*4 + 2][key] = f2bf(vv.z);
                Vt[c0 + j*4 + 3][key] = f2bf(vv.w);
            }
        }
        __syncthreads();

        { // S(16x128): 8 waves, one 16x16 tile each; hd=64 -> 2 wmma k-steps
            v8f s = {};
            #pragma unroll
            for (int j = 0; j < 2; ++j) {
                v8bf al = *(const v8bf*)&Qs[lrow][j * 32 + kb];
                v8bf ah = *(const v8bf*)&Qs[lrow][j * 32 + kb + 16];
                int key = wave * 16 + lrow;
                v8bf bl = *(const v8bf*)&Ks[key][j * 32 + kb];
                v8bf bh = *(const v8bf*)&Ks[key][j * 32 + kb + 16];
                s = wmma_bf16(al, ah, bl, bh, s);
            }
            #pragma unroll
            for (int i = 0; i < 8; ++i)
                Ssh[i + 8 * (lane >> 4)][wave * 16 + lrow] = s[i];
        }
        __syncthreads();

        { // online softmax: 16 threads per row, 8 cols each
            int r = tid >> 4, g = tid & 15;
            float lm = -INFINITY, vals[8];
            #pragma unroll
            for (int i = 0; i < 8; ++i) { vals[i] = Ssh[r][g * 8 + i]; lm = fmaxf(lm, vals[i]); }
            for (int off = 1; off < 16; off <<= 1) lm = fmaxf(lm, __shfl_xor(lm, off, 32));
            float m_new = fmaxf(mrow[r], lm);
            float a = __expf(mrow[r] - m_new);
            float lsum = 0.0f;
            #pragma unroll
            for (int i = 0; i < 8; ++i) {
                float p = __expf(vals[i] - m_new);
                lsum += p;
                Ps[r][g * 8 + i] = f2bf(p);
            }
            for (int off = 1; off < 16; off <<= 1) lsum += __shfl_xor(lsum, off, 32);
            if (g == 0) {
                mrow[r]    = m_new;
                lrow_s[r]  = lrow_s[r] * a + lsum;
                alpha_s[r] = a;
            }
        }
        __syncthreads();

        if (wave < 4) { // O += P(16x128) @ V(128x64); rescale acc by alpha first
            #pragma unroll
            for (int i = 0; i < 8; ++i) acc[i] *= alpha_s[i + 8 * (lane >> 4)];
            #pragma unroll
            for (int j = 0; j < 4; ++j) {
                v8bf al = *(const v8bf*)&Ps[lrow][j * 32 + kb];
                v8bf ah = *(const v8bf*)&Ps[lrow][j * 32 + kb + 16];
                int n = wave * 16 + lrow;
                v8bf bl = *(const v8bf*)&Vt[n][j * 32 + kb];
                v8bf bh = *(const v8bf*)&Vt[n][j * 32 + kb + 16];
                acc = wmma_bf16(al, ah, bl, bh, acc);
            }
        }
        __syncthreads();
    }

    if (wave < 4) { // normalize and write o (bf16, [b,s,h*64+n])
        #pragma unroll
        for (int i = 0; i < 8; ++i) {
            int row = i + 8 * (lane >> 4);
            float v = acc[i] / lrow_s[row];
            int col = wave * 16 + lrow;
            o[((size_t)bi * 2048 + s0 + row) * 1024 + hi * 64 + col] = f2bf(v);
        }
    }
}

// ------------------------------ launcher ------------------------------------
extern "C" void kernel_launch(void* const* d_in, const int* in_sizes, int n_in,
                              void* d_out, int out_size, void* d_ws, size_t ws_size,
                              hipStream_t stream) {
    const int D = 1024, F = 4096, M = 4096;   // M = 2*2048 tokens
    const float* x      = (const float*)d_in[0];
    const float* ln1s   = (const float*)d_in[1];
    const float* ln1b   = (const float*)d_in[2];
    const float* w_qkv  = (const float*)d_in[3];
    const float* b_qkv  = (const float*)d_in[4];
    const float* w_out  = (const float*)d_in[5];
    const float* b_out  = (const float*)d_in[6];
    const float* ln2s   = (const float*)d_in[7];
    const float* ln2b   = (const float*)d_in[8];
    const float* w_fc1  = (const float*)d_in[9];
    const float* b_fc1  = (const float*)d_in[10];
    const float* w_fc2  = (const float*)d_in[11];
    const float* b_fc2  = (const float*)d_in[12];

    char* p = (char*)d_ws;
    auto take = [&](size_t bytes) { char* r = p; p += (bytes + 255) & ~(size_t)255; return r; };
    bf16*  wqkv_t = (bf16*) take((size_t)D * 3 * D * 2);   // [3D][D]
    bf16*  wout_t = (bf16*) take((size_t)D * D * 2);       // [D][D]
    bf16*  wfc1_t = (bf16*) take((size_t)D * F * 2);       // [F][D]
    bf16*  wfc2_t = (bf16*) take((size_t)F * D * 2);       // [D][F]
    bf16*  y1     = (bf16*) take((size_t)M * D * 2);
    float* qkv    = (float*)take((size_t)M * 3 * D * 4);
    bf16*  o_b    = (bf16*) take((size_t)M * D * 2);
    float* x1     = (float*)take((size_t)M * D * 4);
    bf16*  y2     = (bf16*) take((size_t)M * D * 2);
    bf16*  hbuf   = (bf16*) take((size_t)M * F * 2);

    // weights: fp32 [K][N] -> bf16 transposed [N][K]
    cvt_transpose_bf16<<<dim3(3 * D / 32, D / 32), 256, 0, stream>>>(w_qkv, wqkv_t, D, 3 * D);
    cvt_transpose_bf16<<<dim3(D / 32,     D / 32), 256, 0, stream>>>(w_out, wout_t, D, D);
    cvt_transpose_bf16<<<dim3(F / 32,     D / 32), 256, 0, stream>>>(w_fc1, wfc1_t, D, F);
    cvt_transpose_bf16<<<dim3(D / 32,     F / 32), 256, 0, stream>>>(w_fc2, wfc2_t, F, D);

    // attention sublayer
    layernorm_bf16<<<M, 256, 0, stream>>>(x, ln1s, ln1b, y1);
    gemm_wmma<0,0,0><<<dim3(3 * D / 128, M / 128), 256, 0, stream>>>(
        y1, wqkv_t, b_qkv, nullptr, qkv, M, 3 * D, D);
    attention_wmma<<<dim3(2048 / 16, 2 * 16), 256, 0, stream>>>(qkv, o_b);
    gemm_wmma<0,1,0><<<dim3(D / 128, M / 128), 256, 0, stream>>>(
        o_b, wout_t, b_out, x, x1, M, D, D);

    // FFN sublayer
    layernorm_bf16<<<M, 256, 0, stream>>>(x1, ln2s, ln2b, y2);
    gemm_wmma<1,0,1><<<dim3(F / 128, M / 128), 256, 0, stream>>>(
        y2, wfc1_t, b_fc1, nullptr, hbuf, M, F, D);
    gemm_wmma<0,1,0><<<dim3(D / 128, M / 128), 256, 0, stream>>>(
        hbuf, wfc2_t, b_fc2, x1, (float*)d_out, M, D, F);
}
